// HDiTBlock_33157147525234
// MI455X (gfx1250) — matrix operand order repaired and checked by reading, not compiled, not run on hardware
//
#include <hip/hip_runtime.h>
#include <hip/hip_bf16.h>
#include <math.h>

typedef _Float16 v16h __attribute__((ext_vector_type(16)));
typedef _Float16 v8h  __attribute__((ext_vector_type(8)));
typedef float    v8f  __attribute__((ext_vector_type(8)));

#define NTOK   4608   // 2*48*48
#define DMODEL 384
#define NCELL  72     // 2*6*6
#define CROWS  2304   // 768 + 768 + 384 + 384

// ---------------------------------------------------------------------------
// Weight-norm scale: scale[r] = g[r] / ||v_row[r]||  for the 4 v matrices
// rows: [0,768) ada1_v, [768,1536) ada2_v, [1536,1920) gate1_v, [1920,2304) gate2_v
// ---------------------------------------------------------------------------
__global__ __launch_bounds__(128) void wn_scale_kernel(
    const float* __restrict__ v1, const float* __restrict__ g1,
    const float* __restrict__ v2, const float* __restrict__ g2,
    const float* __restrict__ v3, const float* __restrict__ g3,
    const float* __restrict__ v4, const float* __restrict__ g4,
    float* __restrict__ scale)
{
    int rr = blockIdx.x;
    const float* v; const float* g; int r;
    if (rr < 768)        { v = v1 + (size_t)rr * DMODEL;          g = g1; r = rr; }
    else if (rr < 1536)  { v = v2 + (size_t)(rr - 768) * DMODEL;  g = g2; r = rr - 768; }
    else if (rr < 1920)  { v = v3 + (size_t)(rr - 1536) * DMODEL; g = g3; r = rr - 1536; }
    else                 { v = v4 + (size_t)(rr - 1920) * DMODEL; g = g4; r = rr - 1920; }
    int lane = threadIdx.x;
    float s = 0.f;
    for (int c = lane; c < DMODEL; c += 128) s += v[c] * v[c];
    __shared__ float red[128];
    red[lane] = s; __syncthreads();
    for (int o = 64; o > 0; o >>= 1) { if (lane < o) red[lane] += red[lane + o]; __syncthreads(); }
    if (lane == 0) scale[rr] = g[r] * rsqrtf(red[0]);
}

// ---------------------------------------------------------------------------
// silu(cond): 72*384 elements
// ---------------------------------------------------------------------------
__global__ __launch_bounds__(256) void silu_kernel(
    const float* __restrict__ in, float* __restrict__ out, int n)
{
    int i = blockIdx.x * 256 + threadIdx.x;
    if (i < n) { float x = in[i]; out[i] = x / (1.f + __expf(-x)); }
}

// ---------------------------------------------------------------------------
// cond matmul: out[cell][r] = scale[r] * dot(csilu[cell], v_row[r]) ; 72 x 2304
// ---------------------------------------------------------------------------
__global__ __launch_bounds__(256) void condmat_kernel(
    const float* __restrict__ csilu, const float* __restrict__ scale,
    const float* __restrict__ v1, const float* __restrict__ v2,
    const float* __restrict__ v3, const float* __restrict__ v4,
    float* __restrict__ out)
{
    int idx  = blockIdx.x * 256 + threadIdx.x;       // 72*2304 = 165888 exact
    int cell = idx / CROWS, rr = idx % CROWS;
    const float* vp;
    if (rr < 768)       vp = v1 + (size_t)rr * DMODEL;
    else if (rr < 1536) vp = v2 + (size_t)(rr - 768) * DMODEL;
    else if (rr < 1920) vp = v3 + (size_t)(rr - 1536) * DMODEL;
    else                vp = v4 + (size_t)(rr - 1920) * DMODEL;
    const float* cp = csilu + (size_t)cell * DMODEL;
    float s = 0.f;
    for (int k = 0; k < DMODEL; ++k) s = fmaf(cp[k], vp[k], s);
    out[idx] = s * scale[rr];
}

// ---------------------------------------------------------------------------
// f32 -> f16 convert (weights)
// ---------------------------------------------------------------------------
__global__ __launch_bounds__(256) void cvt_f16_kernel(
    const float* __restrict__ in, _Float16* __restrict__ out, int n)
{
    int i = blockIdx.x * 256 + threadIdx.x;
    if (i < n) out[i] = (_Float16)in[i];
}

// ---------------------------------------------------------------------------
// LayerNorm + adaLN modulation, one wave per token (wave32, 12 ch/lane)
// ab stride per cell = 2304; a at [c], b at [384+c]
// ---------------------------------------------------------------------------
__global__ __launch_bounds__(256) void adaln_kernel(
    const float* __restrict__ x, const float* __restrict__ gamma,
    const float* __restrict__ beta, const float* __restrict__ ab,
    _Float16* __restrict__ out)
{
    int t    = blockIdx.x * 8 + (threadIdx.x >> 5);
    int lane = threadIdx.x & 31;
    const float* xp = x + (size_t)t * DMODEL;
    float vals[12];
    float s = 0.f;
    #pragma unroll
    for (int i = 0; i < 12; ++i) { vals[i] = xp[lane + 32 * i]; s += vals[i]; }
    for (int o = 16; o > 0; o >>= 1) s += __shfl_xor(s, o, 32);
    float mu = s * (1.f / DMODEL);
    float vv = 0.f;
    #pragma unroll
    for (int i = 0; i < 12; ++i) { float d = vals[i] - mu; vv += d * d; }
    for (int o = 16; o > 0; o >>= 1) vv += __shfl_xor(vv, o, 32);
    float inv = rsqrtf(vv * (1.f / DMODEL) + 1e-6f);
    int b = t / 2304, rem = t % 2304, hh = rem / 48, ww = rem % 48;
    int cell = b * 36 + (hh >> 3) * 6 + (ww >> 3);
    const float* abp = ab + (size_t)cell * CROWS;
    _Float16* op = out + (size_t)t * DMODEL;
    #pragma unroll
    for (int i = 0; i < 12; ++i) {
        int c = lane + 32 * i;
        float xn = (vals[i] - mu) * inv * gamma[c] + beta[c];
        op[c] = (_Float16)(xn * (1.f + abp[c]) + abp[DMODEL + c]);
    }
}

// ---------------------------------------------------------------------------
// WMMA GEMM, 2x2 register-blocked: each wave computes a 32x32 output tile
// (4 accumulators), loading 2 A frags + 2 B frags per 32-wide K step and
// issuing 4 v_wmma_f32_16x16x32_f16 -> ~16 flops/byte from L2.
// A fragment: lane(r,hi) row=r, k = hi*16 + 0..15 (contiguous, matches B).
// B fragment: lane(r,hi) col=r, k = hi*16 + 0..15 (contiguous along W row).
// MODE 0: store f32. MODE 2: bias+GELU(tanh) -> f16. MODE 3: bias, *gate, +res -> f32
// ---------------------------------------------------------------------------
template <int MODE>
__global__ __launch_bounds__(256) void gemm_wmma_kernel(
    const _Float16* __restrict__ A, const _Float16* __restrict__ W,
    const float* __restrict__ bias,
    float* __restrict__ Cf, _Float16* __restrict__ Ch,
    const float* __restrict__ gate, const float* __restrict__ xres,
    int M, int N, int K)
{
    int wave   = blockIdx.x * 8 + (threadIdx.x >> 5);
    int tilesN = N >> 5;                       // 32-wide tiles
    int tm = wave / tilesN, tn = wave % tilesN;
    if (tm >= (M >> 5)) return;                // uniform per wave -> EXEC stays all-1
    int lane = threadIdx.x & 31;
    int r = lane & 15, hi = lane >> 4;

    const _Float16* ap0 = A + (size_t)(tm * 32 + r) * K + hi * 16;
    const _Float16* ap1 = ap0 + (size_t)16 * K;
    const _Float16* bp0 = W + (size_t)(tn * 32 + r) * K + hi * 16;
    const _Float16* bp1 = bp0 + (size_t)16 * K;

    v8f acc00 = {0.f,0.f,0.f,0.f,0.f,0.f,0.f,0.f};
    v8f acc01 = acc00, acc10 = acc00, acc11 = acc00;

    for (int k0 = 0; k0 < K; k0 += 32) {
        v8h a0lo = *(const v8h*)(ap0 + k0);        // k = k0 + hi*16 .. +7
        v8h a0hi = *(const v8h*)(ap0 + k0 + 8);    // k = k0 + hi*16 + 8 .. +15
        v8h a1lo = *(const v8h*)(ap1 + k0);
        v8h a1hi = *(const v8h*)(ap1 + k0 + 8);
        v16h b0  = *(const v16h*)(bp0 + k0);       // k = k0 + hi*16 .. +15
        v16h b1  = *(const v16h*)(bp1 + k0);
        v16h a0, a1;
        #pragma unroll
        for (int j = 0; j < 8; ++j) {
            a0[j] = a0lo[j]; a0[j + 8] = a0hi[j];
            a1[j] = a1lo[j]; a1[j + 8] = a1hi[j];
        }
        acc00 = __builtin_amdgcn_wmma_f32_16x16x32_f16(false, a0, false, b0, (short)0, acc00, false, false);
        acc01 = __builtin_amdgcn_wmma_f32_16x16x32_f16(false, a0, false, b1, (short)0, acc01, false, false);
        acc10 = __builtin_amdgcn_wmma_f32_16x16x32_f16(false, a1, false, b0, (short)0, acc10, false, false);
        acc11 = __builtin_amdgcn_wmma_f32_16x16x32_f16(false, a1, false, b1, (short)0, acc11, false, false);
    }

    #pragma unroll
    for (int i = 0; i < 2; ++i) {
        #pragma unroll
        for (int j = 0; j < 2; ++j) {
            v8f acc = (i == 0) ? (j == 0 ? acc00 : acc01)
                               : (j == 0 ? acc10 : acc11);
            int col = tn * 32 + j * 16 + r;
            float bval = bias ? bias[col] : 0.f;
            #pragma unroll
            for (int d = 0; d < 8; ++d) {
                int row = tm * 32 + i * 16 + d + 8 * hi;
                float v = acc[d] + bval;
                if (MODE == 0) {
                    Cf[(size_t)row * N + col] = v;
                } else if (MODE == 2) {
                    float t  = 0.7978845608028654f * (v + 0.044715f * v * v * v);
                    float gl = 0.5f * v * (1.f + tanhf(t));
                    Ch[(size_t)row * N + col] = (_Float16)gl;
                } else { // MODE == 3
                    int b = row / 2304, rem = row % 2304, hh = rem / 48, ww = rem % 48;
                    int cell = b * 36 + (hh >> 3) * 6 + (ww >> 3);
                    float gv = gate[(size_t)cell * CROWS + col];
                    Cf[(size_t)row * N + col] = xres[(size_t)row * N + col] + v * gv;
                }
            }
        }
    }
}

// ---------------------------------------------------------------------------
// 7x7 neighborhood attention, one wave per (token, head); head dim 64 = 2/lane
// qkv: [NTOK][1152] f32 (q:0..383, k:384..767, v:768..1151). out: f16 [NTOK][384]
// ---------------------------------------------------------------------------
__global__ __launch_bounds__(256) void natten_kernel(
    const float* __restrict__ qkv, _Float16* __restrict__ out)
{
    int wid  = blockIdx.x * 8 + (threadIdx.x >> 5);  // 0 .. 27647
    int lane = threadIdx.x & 31;
    int t = wid / 6, head = wid % 6;
    int b = t / 2304, rem = t % 2304, h = rem / 48, w = rem % 48;
    int sh = h - 3; sh = sh < 0 ? 0 : (sh > 41 ? 41 : sh);
    int sw = w - 3; sw = sw < 0 ? 0 : (sw > 41 ? 41 : sw);

    const float* qp = qkv + (size_t)t * 1152 + head * 64;
    float q0 = qp[lane], q1 = qp[lane + 32];
    const float scale = 0.125f;   // 1/sqrt(64)

    float logits[49];
    #pragma unroll
    for (int j = 0; j < 49; ++j) {
        int tn = b * 2304 + (sh + j / 7) * 48 + (sw + j % 7);
        const float* kp = qkv + (size_t)tn * 1152 + DMODEL + head * 64;
        float p = q0 * kp[lane] + q1 * kp[lane + 32];
        for (int o = 16; o > 0; o >>= 1) p += __shfl_xor(p, o, 32);
        logits[j] = p * scale;
    }
    float m = logits[0];
    #pragma unroll
    for (int j = 1; j < 49; ++j) m = fmaxf(m, logits[j]);
    float denom = 0.f;
    #pragma unroll
    for (int j = 0; j < 49; ++j) { logits[j] = __expf(logits[j] - m); denom += logits[j]; }
    float rn = 1.f / denom;
    float o0 = 0.f, o1 = 0.f;
    #pragma unroll
    for (int j = 0; j < 49; ++j) {
        int tn = b * 2304 + (sh + j / 7) * 48 + (sw + j % 7);
        const float* vp = qkv + (size_t)tn * 1152 + 2 * DMODEL + head * 64;
        o0 = fmaf(logits[j], vp[lane], o0);
        o1 = fmaf(logits[j], vp[lane + 32], o1);
    }
    _Float16* op = out + (size_t)t * DMODEL + head * 64;
    op[lane]      = (_Float16)(o0 * rn);
    op[lane + 32] = (_Float16)(o1 * rn);
}

// ---------------------------------------------------------------------------
// Launcher
// ---------------------------------------------------------------------------
extern "C" void kernel_launch(void* const* d_in, const int* in_sizes, int n_in,
                              void* d_out, int out_size, void* d_ws, size_t ws_size,
                              hipStream_t stream) {
    const float* x      = (const float*)d_in[0];
    const float* cond   = (const float*)d_in[1];
    const float* ln1_g  = (const float*)d_in[2];
    const float* ln1_b  = (const float*)d_in[3];
    const float* ada1_v = (const float*)d_in[4];
    const float* ada1_g = (const float*)d_in[5];
    const float* ln2_g  = (const float*)d_in[6];
    const float* ln2_b  = (const float*)d_in[7];
    const float* ada2_v = (const float*)d_in[8];
    const float* ada2_g = (const float*)d_in[9];
    const float* gate1_v = (const float*)d_in[10];
    const float* gate1_g = (const float*)d_in[11];
    const float* gate2_v = (const float*)d_in[12];
    const float* gate2_g = (const float*)d_in[13];
    const float* w_qkv  = (const float*)d_in[14];
    const float* b_qkv  = (const float*)d_in[15];
    const float* w_out  = (const float*)d_in[16];
    const float* b_out  = (const float*)d_in[17];
    const float* w_mlp1 = (const float*)d_in[18];
    const float* b_mlp1 = (const float*)d_in[19];
    const float* w_mlp2 = (const float*)d_in[20];
    const float* b_mlp2 = (const float*)d_in[21];

    // workspace carve-up (256B aligned)
    char* ws = (char*)d_ws;
    size_t off = 0;
    auto carve = [&](size_t bytes) -> char* {
        char* p = ws + off;
        off = (off + bytes + 255) & ~(size_t)255;
        return p;
    };
    float*    scale_v  = (float*)   carve((size_t)CROWS * 4);
    float*    csilu    = (float*)   carve((size_t)NCELL * DMODEL * 4);
    float*    cond_out = (float*)   carve((size_t)NCELL * CROWS * 4);
    _Float16* wqkv_h   = (_Float16*)carve((size_t)1152 * 384 * 2);
    _Float16* wout_h   = (_Float16*)carve((size_t)384 * 384 * 2);
    _Float16* wm1_h    = (_Float16*)carve((size_t)1536 * 384 * 2);
    _Float16* wm2_h    = (_Float16*)carve((size_t)384 * 1536 * 2);
    _Float16* h1_h     = (_Float16*)carve((size_t)NTOK * DMODEL * 2);
    float*    qkv_f    = (float*)   carve((size_t)NTOK * 1152 * 4);
    _Float16* attn_h   = (_Float16*)carve((size_t)NTOK * DMODEL * 2);
    float*    x2_f     = (float*)   carve((size_t)NTOK * DMODEL * 4);
    _Float16* h2_h     = (_Float16*)carve((size_t)NTOK * DMODEL * 2);
    _Float16* m1_h     = (_Float16*)carve((size_t)NTOK * 1536 * 2);
    (void)ws_size; (void)in_sizes; (void)n_in; (void)out_size;

    // 1. weight-norm scales + silu(cond) + cond matmuls (72 x 2304)
    wn_scale_kernel<<<CROWS, 128, 0, stream>>>(ada1_v, ada1_g, ada2_v, ada2_g,
                                               gate1_v, gate1_g, gate2_v, gate2_g, scale_v);
    silu_kernel<<<(NCELL * DMODEL) / 256, 256, 0, stream>>>(cond, csilu, NCELL * DMODEL);
    condmat_kernel<<<(NCELL * CROWS) / 256, 256, 0, stream>>>(
        csilu, scale_v, ada1_v, ada2_v, gate1_v, gate2_v, cond_out);

    // 2. weight conversion f32 -> f16
    cvt_f16_kernel<<<(1152 * 384 + 255) / 256, 256, 0, stream>>>(w_qkv, wqkv_h, 1152 * 384);
    cvt_f16_kernel<<<(384 * 384 + 255) / 256, 256, 0, stream>>>(w_out, wout_h, 384 * 384);
    cvt_f16_kernel<<<(1536 * 384 + 255) / 256, 256, 0, stream>>>(w_mlp1, wm1_h, 1536 * 384);
    cvt_f16_kernel<<<(384 * 1536 + 255) / 256, 256, 0, stream>>>(w_mlp2, wm2_h, 384 * 1536);

    // 3. adaLN #1 : x -> h1 (f16)
    adaln_kernel<<<NTOK / 8, 256, 0, stream>>>(x, ln1_g, ln1_b, cond_out + 0, h1_h);

    // 4. qkv GEMM: [4608,384] x [1152,384]^T -> f32   (144*36 = 5184 tiles)
    gemm_wmma_kernel<0><<<(144 * 36) / 8, 256, 0, stream>>>(
        h1_h, wqkv_h, b_qkv, qkv_f, nullptr, nullptr, nullptr, NTOK, 1152, DMODEL);

    // 5. neighborhood attention -> attn (f16)
    natten_kernel<<<(NTOK * 6) / 8, 256, 0, stream>>>(qkv_f, attn_h);

    // 6. out-proj GEMM + gate1 + residual -> x2 (f32)   (144*12 tiles)
    gemm_wmma_kernel<3><<<(144 * 12) / 8, 256, 0, stream>>>(
        attn_h, wout_h, b_out, x2_f, nullptr, cond_out + 1536, x, NTOK, DMODEL, DMODEL);

    // 7. adaLN #2 : x2 -> h2 (f16)
    adaln_kernel<<<NTOK / 8, 256, 0, stream>>>(x2_f, ln2_g, ln2_b, cond_out + 768, h2_h);

    // 8. mlp1 GEMM + GELU -> m1 (f16)   (144*48 tiles)
    gemm_wmma_kernel<2><<<(144 * 48) / 8, 256, 0, stream>>>(
        h2_h, wm1_h, b_mlp1, nullptr, m1_h, nullptr, nullptr, NTOK, 1536, DMODEL);

    // 9. mlp2 GEMM + gate2 + residual -> d_out (f32)   (144*12 tiles)
    gemm_wmma_kernel<3><<<(144 * 12) / 8, 256, 0, stream>>>(
        m1_h, wm2_h, b_mlp2, (float*)d_out, nullptr, cond_out + 1920, x2_f,
        NTOK, DMODEL, 1536);
}
